// CausalMultiHeadSelfAttention_45122926412045
// MI455X (gfx1250) — matrix operand order repaired and checked by reading, not compiled, not run on hardware
//
#include <hip/hip_runtime.h>
#include <hip/hip_bf16.h>

// ---------------------------------------------------------------------------
// CDNA5 (gfx1250) types
// ---------------------------------------------------------------------------
typedef __bf16 bf16_t;
typedef __attribute__((ext_vector_type(16))) __bf16 v16bf;
typedef __attribute__((ext_vector_type(8)))  __bf16 v8bf;
typedef __attribute__((ext_vector_type(8)))  float  v8f;
typedef __attribute__((ext_vector_type(4)))  int    v4i;

#define B_SZ   2
#define S_SZ   2048
#define D_SZ   1024
#define H_SZ   16
#define DH_SZ  64
#define M_SZ   (B_SZ * S_SZ)      // 4096 rows
#define NEG_BIG (-1.0e9f)

// ---- optional CDNA5 async global->LDS copy path --------------------------
#define USE_ASYNC_LDS 0
#if defined(__has_builtin)
#if __has_builtin(__builtin_amdgcn_global_load_async_to_lds_b128)
#undef USE_ASYNC_LDS
#define USE_ASYNC_LDS 1
#endif
#endif

#if USE_ASYNC_LDS
typedef __attribute__((address_space(1))) v4i* g4p;   // global int4*
typedef __attribute__((address_space(3))) v4i* l4p;   // LDS int4*
__device__ __forceinline__ void async_cp16(const void* g, void* l) {
  // prototype (from clang diagnostic): (int4 AS1*, int4 AS3*, imm, imm)
  __builtin_amdgcn_global_load_async_to_lds_b128(
      (g4p)(v4i*)g, (l4p)(v4i*)l, 0, 0);
}
#endif

__device__ __forceinline__ bf16_t f2bf(float f) {
  // native RNE convert; lowers to v_cvt_pk_bf16_f32 pairs on gfx1250
  return static_cast<bf16_t>(f);
}

__device__ __forceinline__ v8f wmma_bf16(v16bf a, v16bf b, v8f c) {
  // D = A(16x32 bf16) x B(32x16 bf16) + C(16x16 f32)
  return __builtin_amdgcn_wmma_f32_16x16x32_bf16(
      false, a, false, b, (short)0, c, false, false);
}

// A-fragment (16x32 bf16), row-major source, leading dim ld.
// lane<16: row=lane, K={kb..kb+7, kb+16..kb+23}, kb=0; lane>=16: kb=8.
__device__ __forceinline__ v16bf load_a_frag(const bf16_t* base, int row_base,
                                             int ld, int k0) {
  int lane = threadIdx.x & 31;
  int m  = row_base + (lane & 15);
  int kb = (lane & 16) ? 8 : 0;
  const bf16_t* p = base + (size_t)m * ld + k0 + kb;
  v8bf lo = *reinterpret_cast<const v8bf*>(p);
  v8bf hi = *reinterpret_cast<const v8bf*>(p + 16);
  v16bf out;
#pragma unroll
  for (int i = 0; i < 8; ++i) { out[i] = lo[i]; out[i + 8] = hi[i]; }
  return out;
}

// B-fragment (32x16 bf16): B[k][n] = Src[col_base+n][k0+k], Src row-major, ld.
// lane&15 = column n; lanes<16 hold K={0..7,16..23}, lanes>=16 hold K={8..15,24..31}.
__device__ __forceinline__ v16bf load_b_frag(const bf16_t* base, int col_base,
                                             int ld, int k0) {
  int lane = threadIdx.x & 31;
  int n    = lane & 15;
  int koff = (lane & 16) ? 8 : 0;
  const bf16_t* p = base + (size_t)(col_base + n) * ld + k0 + koff;
  v8bf lo = *reinterpret_cast<const v8bf*>(p);
  v8bf hi = *reinterpret_cast<const v8bf*>(p + 16);
  v16bf out;
#pragma unroll
  for (int i = 0; i < 8; ++i) { out[i] = lo[i]; out[i + 8] = hi[i]; }
  return out;
}

// C/D f32 16x16 store: VGPR r holds row (r + 8*(lane>=16)), col = lane&15.
__device__ __forceinline__ void store_tile_f32(float* C, int ldc, int rb,
                                               int cb, v8f c) {
  int lane = threadIdx.x & 31;
  int n  = lane & 15;
  int mh = (lane & 16) ? 8 : 0;
#pragma unroll
  for (int r = 0; r < 8; ++r)
    C[(size_t)(rb + mh + r) * ldc + cb + n] = c[r];
}

// ---------------------------------------------------------------------------
// fp32 -> bf16 cast
// ---------------------------------------------------------------------------
__global__ __launch_bounds__(256) void cast_bf16_kernel(const float* __restrict__ in,
                                                        bf16_t* __restrict__ out,
                                                        int n) {
  int i = blockIdx.x * 256 + threadIdx.x;
  if (i < n) out[i] = f2bf(in[i]);
}

// ---------------------------------------------------------------------------
// C[M,N] = A[M,K](bf16) * W[N,K]^T(bf16), fp32 out.
// Block = 256 thr (8 waves), block tile 128x128; wave tile 32x64 (2x4 wmma).
// ---------------------------------------------------------------------------
__global__ __launch_bounds__(256) void gemm_bf16_nt(const bf16_t* __restrict__ A,
                                                    const bf16_t* __restrict__ W,
                                                    float* __restrict__ C,
                                                    int K, int N) {
  int wave = threadIdx.x >> 5;
  int wm = wave >> 1;          // 0..3
  int wn = wave & 1;           // 0..1
  int rb = blockIdx.y * 128 + wm * 32;
  int cb = blockIdx.x * 128 + wn * 64;

  v8f c00 = {}, c01 = {}, c02 = {}, c03 = {};
  v8f c10 = {}, c11 = {}, c12 = {}, c13 = {};
  for (int k = 0; k < K; k += 32) {
    v16bf a0 = load_a_frag(A, rb,      K, k);
    v16bf a1 = load_a_frag(A, rb + 16, K, k);
    v16bf b0 = load_b_frag(W, cb,      K, k);
    v16bf b1 = load_b_frag(W, cb + 16, K, k);
    v16bf b2 = load_b_frag(W, cb + 32, K, k);
    v16bf b3 = load_b_frag(W, cb + 48, K, k);
    c00 = wmma_bf16(a0, b0, c00);
    c01 = wmma_bf16(a0, b1, c01);
    c02 = wmma_bf16(a0, b2, c02);
    c03 = wmma_bf16(a0, b3, c03);
    c10 = wmma_bf16(a1, b0, c10);
    c11 = wmma_bf16(a1, b1, c11);
    c12 = wmma_bf16(a1, b2, c12);
    c13 = wmma_bf16(a1, b3, c13);
  }
  store_tile_f32(C, N, rb,      cb,      c00);
  store_tile_f32(C, N, rb,      cb + 16, c01);
  store_tile_f32(C, N, rb,      cb + 32, c02);
  store_tile_f32(C, N, rb,      cb + 48, c03);
  store_tile_f32(C, N, rb + 16, cb,      c10);
  store_tile_f32(C, N, rb + 16, cb + 16, c11);
  store_tile_f32(C, N, rb + 16, cb + 32, c12);
  store_tile_f32(C, N, rb + 16, cb + 48, c13);
}

// ---------------------------------------------------------------------------
// RoPE + head-split: X f32 [B*S, 1024] -> Out bf16 [B,H,S,Dh].
// ---------------------------------------------------------------------------
__global__ __launch_bounds__(256) void rope_split_kernel(const float* __restrict__ X,
                                                         bf16_t* __restrict__ Out) {
  int idx = blockIdx.x * 256 + threadIdx.x;   // M*H*32 threads
  int i = idx & 31;
  int h = (idx >> 5) & 15;
  int m = idx >> 9;                            // 0..M-1
  int b = m >> 11;                             // /S_SZ
  int s = m & (S_SZ - 1);
  float inv = __expf(-(float)(2 * i) * 0.14391157f);  // ln(10000)/64
  float ang = (float)s * inv;
  float cs = cosf(ang), sn = sinf(ang);
  const float* xp = X + (size_t)m * D_SZ + h * DH_SZ;
  float x1 = xp[2 * i], x2 = xp[2 * i + 1];
  bf16_t* op = Out + (((size_t)(b * H_SZ + h) * S_SZ) + s) * DH_SZ;
  op[2 * i]     = f2bf(x1 * cs - x2 * sn);
  op[2 * i + 1] = f2bf(x1 * sn + x2 * cs);
}

// V: cast + transpose to [B,H,Dh,S].
__global__ __launch_bounds__(256) void v_cast_t_kernel(const float* __restrict__ X,
                                                       bf16_t* __restrict__ Vt) {
  int idx = blockIdx.x * 256 + threadIdx.x;   // M*D threads
  int d = idx & 63;
  int h = (idx >> 6) & 15;
  int m = idx >> 10;
  int b = m >> 11;
  int s = m & (S_SZ - 1);
  Vt[(((size_t)(b * H_SZ + h) * DH_SZ) + d) * S_SZ + s] =
      f2bf(X[(size_t)m * D_SZ + h * DH_SZ + d]);
}

// ---------------------------------------------------------------------------
// Causal flash attention.
// Block = 4 waves, 128 consecutive q rows of one (b,h); wave owns 32 q rows.
// K/V staged per 32-key block into LDS cooperatively (async copy when
// available, double-buffered), shared by all 4 waves.
// ---------------------------------------------------------------------------

// B-fragment from LDS-resident matrix, same layout rules as load_b_frag.
__device__ __forceinline__ v16bf load_b_frag_lds(const bf16_t* base, int col_base,
                                                 int ld, int k0) {
  int lane = threadIdx.x & 31;
  int n    = lane & 15;
  int koff = (lane & 16) ? 8 : 0;
  const bf16_t* p = base + (col_base + n) * ld + k0 + koff;
  v8bf lo = *reinterpret_cast<const v8bf*>(p);
  v8bf hi = *reinterpret_cast<const v8bf*>(p + 16);
  v16bf out;
#pragma unroll
  for (int i = 0; i < 8; ++i) { out[i] = lo[i]; out[i + 8] = hi[i]; }
  return out;
}

// Stage one 32-key block: K (32 keys x 64 d, row-major) and Vt (64 d x 32 keys).
// 128 threads x 4 chunks of 16B (2 for K, 2 for V).
__device__ __forceinline__ void stage_kv(const bf16_t* __restrict__ Kp,
                                         const bf16_t* __restrict__ Vp,
                                         bf16_t* Ks, bf16_t* Vs,
                                         int kb, int tid) {
#pragma unroll
  for (int half = 0; half < 2; ++half) {
    int ck = tid + half * 128;               // K chunk: row=ck>>3, 16B chunk ck&7
    const bf16_t* gk = Kp + (size_t)(kb + (ck >> 3)) * DH_SZ + (ck & 7) * 8;
    bf16_t* lk = Ks + (ck >> 3) * DH_SZ + (ck & 7) * 8;
    int cv = tid + half * 128;               // V chunk: row(d)=cv>>2, chunk cv&3
    const bf16_t* gv = Vp + (size_t)(cv >> 2) * S_SZ + kb + (cv & 3) * 8;
    bf16_t* lv = Vs + (cv >> 2) * 32 + (cv & 3) * 8;
#if USE_ASYNC_LDS
    async_cp16(gk, lk);
    async_cp16(gv, lv);
#else
    *reinterpret_cast<uint4*>(lk) = *reinterpret_cast<const uint4*>(gk);
    *reinterpret_cast<uint4*>(lv) = *reinterpret_cast<const uint4*>(gv);
#endif
  }
}

__global__ __launch_bounds__(128) void flash_attn_kernel(const bf16_t* __restrict__ Q,
                                                         const bf16_t* __restrict__ Km,
                                                         const bf16_t* __restrict__ Vt,
                                                         bf16_t* __restrict__ O) {
  __shared__ bf16_t Ks[2][32 * DH_SZ];   // [buf][key][d]      4KB each
  __shared__ bf16_t Vs[2][DH_SZ * 32];   // [buf][d][key]      4KB each
  __shared__ float  Pl[4][32][36];       // per-wave P staging (32q x 32k)

  int tid  = threadIdx.x;
  int lane = tid & 31;
  int wave = tid >> 5;
  const int nqb = S_SZ / 128;            // 16 q-blocks per (b,h)
  int qb0 = (blockIdx.x & (nqb - 1)) * 128;
  int bh  = blockIdx.x >> 4;             // / nqb
  int wqbase = qb0 + wave * 32;          // this wave's 32 q rows

  const bf16_t* Qp = Q  + (size_t)bh * S_SZ * DH_SZ;
  const bf16_t* Kp = Km + (size_t)bh * S_SZ * DH_SZ;
  const bf16_t* Vp = Vt + (size_t)bh * DH_SZ * S_SZ;

  // Q tiles as A-fragments (constant across key loop)
  v16bf q00 = load_a_frag(Qp, wqbase,      DH_SZ, 0);
  v16bf q01 = load_a_frag(Qp, wqbase,      DH_SZ, 32);
  v16bf q10 = load_a_frag(Qp, wqbase + 16, DH_SZ, 0);
  v16bf q11 = load_a_frag(Qp, wqbase + 16, DH_SZ, 32);

  v8f a00 = {}, a01 = {}, a02 = {}, a03 = {};   // O rows 0-15
  v8f a10 = {}, a11 = {}, a12 = {}, a13 = {};   // O rows 16-31
  float rmax[16], rsum[16];
#pragma unroll
  for (int r = 0; r < 16; ++r) { rmax[r] = -3.0e38f; rsum[r] = 0.0f; }

  const float scale = 0.125f;            // 1/sqrt(64)
  int ln = lane & 15;
  int mh = (lane & 16) ? 8 : 0;

  int nblk = (qb0 + 128) >> 5;           // 32-key blocks needed by this block
  stage_kv(Kp, Vp, Ks[0], Vs[0], 0, tid);

  for (int i = 0; i < nblk; ++i) {
    int kb = i << 5;
#if USE_ASYNC_LDS
    asm volatile("s_wait_asynccnt 0x0" ::: "memory");
#endif
    __syncthreads();                     // buf[i&1] ready; prior buffer free
    if (i + 1 < nblk)
      stage_kv(Kp, Vp, Ks[(i + 1) & 1], Vs[(i + 1) & 1], kb + 32, tid);

    if (kb < wqbase + 32) {              // causal: this wave still has work
      const bf16_t* KsB = Ks[i & 1];
      const bf16_t* VsB = Vs[i & 1];

      // ---- scores: 4 tiles (2 q-tiles x 2 key-halves), K frags shared ----
      v16bf kA0 = load_b_frag_lds(KsB, 0,  DH_SZ, 0);
      v16bf kA1 = load_b_frag_lds(KsB, 0,  DH_SZ, 32);
      v16bf kB0 = load_b_frag_lds(KsB, 16, DH_SZ, 0);
      v16bf kB1 = load_b_frag_lds(KsB, 16, DH_SZ, 32);
      v8f s00 = {}, s01 = {}, s10 = {}, s11 = {};
      s00 = wmma_bf16(q00, kA0, s00); s00 = wmma_bf16(q01, kA1, s00);
      s01 = wmma_bf16(q00, kB0, s01); s01 = wmma_bf16(q01, kB1, s01);
      s10 = wmma_bf16(q10, kA0, s10); s10 = wmma_bf16(q11, kA1, s10);
      s11 = wmma_bf16(q10, kB0, s11); s11 = wmma_bf16(q11, kB1, s11);

      // ---- online softmax, both 16-row groups ----
#pragma unroll
      for (int t = 0; t < 2; ++t) {
        v8f& sa = t ? s10 : s00;
        v8f& sb = t ? s11 : s01;
        v8f* ac[4] = { t ? &a10 : &a00, t ? &a11 : &a01,
                       t ? &a12 : &a02, t ? &a13 : &a03 };
#pragma unroll
        for (int r = 0; r < 8; ++r) {
          int j = t * 8 + r;
          int q = wqbase + t * 16 + mh + r;
          float sc0 = (kb + ln      <= q) ? sa[r] * scale : NEG_BIG;
          float sc1 = (kb + 16 + ln <= q) ? sb[r] * scale : NEG_BIG;
          float mx = fmaxf(sc0, sc1);
          mx = fmaxf(mx, __shfl_xor(mx, 1));
          mx = fmaxf(mx, __shfl_xor(mx, 2));
          mx = fmaxf(mx, __shfl_xor(mx, 4));
          mx = fmaxf(mx, __shfl_xor(mx, 8));   // within 16-lane row group
          float mnew = fmaxf(rmax[j], mx);
          float corr = __expf(rmax[j] - mnew);
          float p0 = __expf(sc0 - mnew);
          float p1 = __expf(sc1 - mnew);
          float ps = p0 + p1;
          ps += __shfl_xor(ps, 1);
          ps += __shfl_xor(ps, 2);
          ps += __shfl_xor(ps, 4);
          ps += __shfl_xor(ps, 8);
          rsum[j] = rsum[j] * corr + ps;
          rmax[j] = mnew;
          (*ac[0])[r] *= corr; (*ac[1])[r] *= corr;
          (*ac[2])[r] *= corr; (*ac[3])[r] *= corr;
          Pl[wave][t * 16 + mh + r][ln]      = p0;
          Pl[wave][t * 16 + mh + r][16 + ln] = p1;
        }
      }

      // ---- re-layout P rows into A-fragments (f32 -> bf16) ----
      int am  = ln;
      int akb = (lane & 16) ? 8 : 0;
      v16bf pf0, pf1;
#pragma unroll
      for (int ii = 0; ii < 8; ++ii) {
        pf0[ii]     = f2bf(Pl[wave][am][akb + ii]);
        pf0[ii + 8] = f2bf(Pl[wave][am][akb + 16 + ii]);
        pf1[ii]     = f2bf(Pl[wave][16 + am][akb + ii]);
        pf1[ii + 8] = f2bf(Pl[wave][16 + am][akb + 16 + ii]);
      }

      // ---- O += P * V, V frags shared by both P tiles ----
      v16bf v0 = load_b_frag_lds(VsB, 0,  32, 0);
      v16bf v1 = load_b_frag_lds(VsB, 16, 32, 0);
      v16bf v2 = load_b_frag_lds(VsB, 32, 32, 0);
      v16bf v3 = load_b_frag_lds(VsB, 48, 32, 0);
      a00 = wmma_bf16(pf0, v0, a00); a01 = wmma_bf16(pf0, v1, a01);
      a02 = wmma_bf16(pf0, v2, a02); a03 = wmma_bf16(pf0, v3, a03);
      a10 = wmma_bf16(pf1, v0, a10); a11 = wmma_bf16(pf1, v1, a11);
      a12 = wmma_bf16(pf1, v2, a12); a13 = wmma_bf16(pf1, v3, a13);
    }
  }

  // ---- epilogue: normalize, write [B*S, 1024] bf16 (A operand for Wo) ----
  int b = bh >> 4, h = bh & 15;
#pragma unroll
  for (int t = 0; t < 2; ++t) {
    size_t rowbase = (size_t)b * S_SZ + wqbase + t * 16;
    v8f* ac[4] = { t ? &a10 : &a00, t ? &a11 : &a01,
                   t ? &a12 : &a02, t ? &a13 : &a03 };
#pragma unroll
    for (int r = 0; r < 8; ++r) {
      float inv = 1.0f / rsum[t * 8 + r];
      bf16_t* op = O + (rowbase + mh + r) * D_SZ + h * DH_SZ;
      op[0  + ln] = f2bf((*ac[0])[r] * inv);
      op[16 + ln] = f2bf((*ac[1])[r] * inv);
      op[32 + ln] = f2bf((*ac[2])[r] * inv);
      op[48 + ln] = f2bf((*ac[3])[r] * inv);
    }
  }
}

// ---------------------------------------------------------------------------
// Host-side orchestration (all on `stream`; graph-capture safe).
// Workspace layout (64 MiB total):
//   xb 8M | Wqb/Wkb/Wvb/Wob 2M each | tmpf 16M | Qb 8M | Kb 8M | Vtb 8M | Ob 8M
// ---------------------------------------------------------------------------
extern "C" void kernel_launch(void* const* d_in, const int* in_sizes, int n_in,
                              void* d_out, int out_size, void* d_ws, size_t ws_size,
                              hipStream_t stream) {
  (void)in_sizes; (void)n_in; (void)out_size; (void)ws_size;
  const float* x  = (const float*)d_in[0];
  const float* Wq = (const float*)d_in[1];
  const float* Wk = (const float*)d_in[2];
  const float* Wv = (const float*)d_in[3];
  const float* Wo = (const float*)d_in[4];
  float* out = (float*)d_out;

  char* ws = (char*)d_ws;
  bf16_t* xb  = (bf16_t*)ws;  ws += (size_t)M_SZ * D_SZ * 2;   // 8 MiB
  bf16_t* Wqb = (bf16_t*)ws;  ws += (size_t)D_SZ * D_SZ * 2;   // 2 MiB
  bf16_t* Wkb = (bf16_t*)ws;  ws += (size_t)D_SZ * D_SZ * 2;
  bf16_t* Wvb = (bf16_t*)ws;  ws += (size_t)D_SZ * D_SZ * 2;
  bf16_t* Wob = (bf16_t*)ws;  ws += (size_t)D_SZ * D_SZ * 2;
  float*  tmpf = (float*)ws;  ws += (size_t)M_SZ * D_SZ * 4;   // 16 MiB
  bf16_t* Qb  = (bf16_t*)ws;  ws += (size_t)M_SZ * D_SZ * 2;   // 8 MiB
  bf16_t* Kb  = (bf16_t*)ws;  ws += (size_t)M_SZ * D_SZ * 2;
  bf16_t* Vtb = (bf16_t*)ws;  ws += (size_t)M_SZ * D_SZ * 2;
  bf16_t* Ob  = (bf16_t*)ws;  ws += (size_t)M_SZ * D_SZ * 2;

  const int nx = M_SZ * D_SZ;        // 4,194,304
  const int nw = D_SZ * D_SZ;        // 1,048,576
  cast_bf16_kernel<<<(nx + 255) / 256, 256, 0, stream>>>(x,  xb,  nx);
  cast_bf16_kernel<<<(nw + 255) / 256, 256, 0, stream>>>(Wq, Wqb, nw);
  cast_bf16_kernel<<<(nw + 255) / 256, 256, 0, stream>>>(Wk, Wkb, nw);
  cast_bf16_kernel<<<(nw + 255) / 256, 256, 0, stream>>>(Wv, Wvb, nw);
  cast_bf16_kernel<<<(nw + 255) / 256, 256, 0, stream>>>(Wo, Wob, nw);

  dim3 gg(D_SZ / 128, M_SZ / 128);   // (8, 32)
  const int nrope = M_SZ * H_SZ * 32;

  // Q projection + RoPE
  gemm_bf16_nt<<<gg, 256, 0, stream>>>(xb, Wqb, tmpf, D_SZ, D_SZ);
  rope_split_kernel<<<nrope / 256, 256, 0, stream>>>(tmpf, Qb);
  // K projection + RoPE
  gemm_bf16_nt<<<gg, 256, 0, stream>>>(xb, Wkb, tmpf, D_SZ, D_SZ);
  rope_split_kernel<<<nrope / 256, 256, 0, stream>>>(tmpf, Kb);
  // V projection + transpose
  gemm_bf16_nt<<<gg, 256, 0, stream>>>(xb, Wvb, tmpf, D_SZ, D_SZ);
  v_cast_t_kernel<<<nx / 256, 256, 0, stream>>>(tmpf, Vtb);

  // Causal flash attention: B*H*(S/128) blocks of 4 waves
  flash_attn_kernel<<<B_SZ * H_SZ * (S_SZ / 128), 128, 0, stream>>>(Qb, Kb, Vtb, Ob);

  // Output projection straight into d_out (fp32)
  gemm_bf16_nt<<<gg, 256, 0, stream>>>(Ob, Wob, out, D_SZ, D_SZ);
}
